// SNNSatelliteDetector_24489903522156
// MI455X (gfx1250) — matrix operand order repaired and checked by reading, not compiled, not run on hardware
//
#include <hip/hip_runtime.h>

typedef __attribute__((ext_vector_type(16))) _Float16 v16h;
typedef __attribute__((ext_vector_type(8)))  float    v8f;

#define LIF_BETA 0.9f
#define BN_EPS   1e-5f

// ---------------------------------------------------------------------------
// Pack OIHW f32 weights into the CDNA5 WMMA A-matrix (16x32 f16) lane layout.
// Loader reads: wpk[((mt*nkc + kc)*32 + lane)*16 + j]
// lane<16 : M=lane,    slot j -> K = kc*32 + (j<8 ? j : j+8)       (K 0..7,16..23)
// lane>=16: M=lane-16, slot j -> K = kc*32 + 8 + (j<8 ? j : j+8)   (K 8..15,24..31)
// ---------------------------------------------------------------------------
__global__ void pack_weights(const float* __restrict__ w, _Float16* __restrict__ wpk,
                             int Cout, int nK, int nkc) {
  int tid = blockIdx.x * blockDim.x + threadIdx.x;
  int total = ((Cout + 15) / 16) * nkc * 512;
  if (tid >= total) return;
  int j    = tid & 15;
  int lane = (tid >> 4) & 31;
  int blk  = tid >> 9;
  int kc   = blk % nkc;
  int mt   = blk / nkc;
  int m = mt * 16 + (lane & 15);
  int k = kc * 32 + ((lane < 16) ? 0 : 8) + ((j < 8) ? j : j + 8);
  float v = 0.f;
  if (m < Cout && k < nK) v = w[(size_t)m * nK + k];
  wpk[tid] = (_Float16)v;
}

// ---------------------------------------------------------------------------
// Implicit-GEMM conv on v_wmma_f32_16x16x32_f16.
// One wave owns MT M-tiles (16 Cout rows each) x one 16-pixel N-tile:
// per K-chunk it gathers ONE shared B fragment and issues MT WMMAs.
// LDS_W=true: the block's packed-A weight slab (contiguous, MT*nkc*512 f16)
// is staged cooperatively into LDS once, then A fragments come from
// ds_load_b128 — weights are fetched from global once per block instead of
// once per wave. 4 waves / block on adjacent pixel tiles.
// TIN = float (dense features) or _Float16 (spike maps, scale elided).
// ---------------------------------------------------------------------------
template <int KHW, int MT, bool LDS_W, typename TIN>
__global__ void __launch_bounds__(128)
conv_wmma(const TIN* __restrict__ in, const _Float16* __restrict__ wpk,
          const float* __restrict__ bias_, float* __restrict__ out,
          int B, int Cin, int Hi, int Wi, long in_bstride,
          int Cout, int Ho, int Wo, int stride, int pad,
          float in_scale, int relu_flag) {
  extern __shared__ char lds_raw[];
  _Float16* lds_a = (_Float16*)lds_raw;

  const int  lane  = threadIdx.x & 31;
  const long Ntot  = (long)B * Ho * Wo;
  const long ntile = (long)blockIdx.x * 4 + (threadIdx.x >> 5);

  constexpr int kk2 = KHW * KHW;
  const int nK     = Cin * kk2;
  const int nkc    = (nK + 31) >> 5;
  const int mtiles = (Cout + 15) >> 4;
  const int mbase  = blockIdx.y * MT;

  if (LDS_W) {
    // Cooperative global->LDS stage of this block's weight slab (before any
    // exit so every thread reaches the barrier).
    const int slab_words = MT * nkc * 256;  // uint32 words (512 f16 per chunk)
    const unsigned int* __restrict__ gsrc =
        (const unsigned int*)(wpk + (size_t)mbase * nkc * 512);
    unsigned int* ldst = (unsigned int*)lds_a;
    for (int i = threadIdx.x; i < slab_words; i += 128) ldst[i] = gsrc[i];
    __syncthreads();
  }
  if (ntile * 16 >= Ntot) return;               // wave-uniform exit

  // This lane supplies B-matrix column N = lane & 15 of the tile.
  const long ncol = ntile * 16 + (lane & 15);
  const bool ncol_ok = ncol < Ntot;
  int b = 0, oy = 0, ox = 0;
  if (ncol_ok) {
    long r = ncol;
    b = (int)(r / ((long)Ho * Wo)); r -= (long)b * Ho * Wo;
    oy = (int)(r / Wo); ox = (int)(r % Wo);
  }
  const TIN* __restrict__ inb = in + (long)b * in_bstride;
  const int iy0 = oy * stride - pad, ix0 = ox * stride - pad;
  const int kbaseB = (lane < 16) ? 0 : 8;       // B layout: K 0..7,16..23 / 8..15,24..31

  v8f acc[MT];
#pragma unroll
  for (int mt = 0; mt < MT; ++mt) acc[mt] = (v8f){};

  for (int kc = 0; kc < nkc; ++kc) {
    // ---- shared B fragment (im2col gather), built once per K-chunk ----
    v16h bv = {};
    if (ncol_ok) {
      const int kb = kc * 32 + kbaseB;
#pragma unroll
      for (int j = 0; j < 16; ++j) {
        float val = 0.f;
        int k = kb + ((j < 8) ? j : j + 8);
        if (k < nK) {
          int cin = k / kk2;
          int rem = k - cin * kk2;
          int r = rem / KHW, s = rem - (rem / KHW) * KHW;
          int iy = iy0 + r, ix = ix0 + s;
          if ((unsigned)iy < (unsigned)Hi && (unsigned)ix < (unsigned)Wi) {
            val = (float)inb[(size_t)cin * Hi * Wi + (size_t)iy * Wi + ix];
            if (__is_same(TIN, float)) val *= in_scale;  // spikes: scale==1
          }
        }
        bv[j] = (_Float16)val;
      }
    }
    // ---- MT A fragments (LDS or global) + MT WMMAs sharing bv ----
#pragma unroll
    for (int mt = 0; mt < MT; ++mt) {
      v16h av = {};
      if (LDS_W) {
        av = *(const v16h*)(lds_a + ((size_t)(mt * nkc + kc) * 32 + lane) * 16);
      } else {
        const int mtile = mbase + mt;
        if (mtile < mtiles)
          av = *(const v16h*)(wpk + ((size_t)(mtile * nkc + kc) * 32 + lane) * 16);
        if (mt == 0 && kc + 1 < nkc)  // gfx1250 global_prefetch for next A chunk
          __builtin_prefetch(wpk + ((size_t)(mbase * nkc + kc + 1) * 32 + lane) * 16, 0, 3);
      }
      acc[mt] = __builtin_amdgcn_wmma_f32_16x16x32_f16(
          false, av, false, bv, (short)0, acc[mt], false, false);
    }
  }

  if (!ncol_ok) return;
  // D layout: VGPR r holds M = r (lanes 0-15) / r+8 (lanes 16-31), N = lane%16
  const size_t opix = (size_t)oy * Wo + ox;
#pragma unroll
  for (int mt = 0; mt < MT; ++mt) {
#pragma unroll
    for (int r = 0; r < 8; ++r) {
      int m = (mbase + mt) * 16 + r + ((lane < 16) ? 0 : 8);
      if (m < Cout) {
        float v = acc[mt][r];
        if (bias_) v += bias_[m];
        if (relu_flag) v = fmaxf(v, 0.f);
        out[((size_t)b * Cout + m) * ((size_t)Ho * Wo) + opix] = v;
      }
    }
  }
}

template <typename TIN>
static void conv_dispatch(const TIN* in, const _Float16* wpk, const float* bias_,
                          float* outp, int B, int Cin, int Hi, int Wi, long bstr,
                          int Cout, int Ho, int Wo, int KHW, int stride, int pad,
                          float scale, int relu, hipStream_t stream) {
  long Ntot = (long)B * Ho * Wo;
  int mtiles = (Cout + 15) / 16;
  int MT = (mtiles >= 4) ? 4 : ((mtiles >= 2) ? 2 : 1);
  int nkc = (Cin * KHW * KHW + 31) / 32;
  size_t slab = (size_t)MT * nkc * 1024;        // bytes of packed A per block
  bool useLds = slab <= (size_t)96 * 1024;      // keep >=3 blocks/WGP otherwise
  size_t shmem = useLds ? slab : 0;
  dim3 grid((unsigned)((Ntot + 63) / 64), (unsigned)((mtiles + MT - 1) / MT), 1);

#define LAUNCH_CONV(KHW_, MT_, LDSW_)                                          \
  conv_wmma<KHW_, MT_, LDSW_, TIN><<<grid, 128, shmem, stream>>>(              \
      in, wpk, bias_, outp, B, Cin, Hi, Wi, bstr, Cout, Ho, Wo, stride, pad,   \
      scale, relu)

  if (KHW == 3) {
    if (MT == 4)      { if (useLds) LAUNCH_CONV(3, 4, true); else LAUNCH_CONV(3, 4, false); }
    else if (MT == 2) { if (useLds) LAUNCH_CONV(3, 2, true); else LAUNCH_CONV(3, 2, false); }
    else              { if (useLds) LAUNCH_CONV(3, 1, true); else LAUNCH_CONV(3, 1, false); }
  } else {
    if (MT == 4)      { if (useLds) LAUNCH_CONV(1, 4, true); else LAUNCH_CONV(1, 4, false); }
    else if (MT == 2) { if (useLds) LAUNCH_CONV(1, 2, true); else LAUNCH_CONV(1, 2, false); }
    else              { if (useLds) LAUNCH_CONV(1, 1, true); else LAUNCH_CONV(1, 1, false); }
  }
#undef LAUNCH_CONV
}

// ---------------------------------------------------------------------------
// Training-mode BN stats: one block per channel; emits fused scale/shift.
// scale = g * rsqrt(var + eps); shift = b - mean * scale
// ---------------------------------------------------------------------------
__global__ void bn_stats(const float* __restrict__ cur, const float* __restrict__ g,
                         const float* __restrict__ beta, float* __restrict__ ss,
                         int C, int B, long chw) {
  const int c = blockIdx.x;
  const long bstride = (long)C * chw;
  const long N = (long)B * chw;
  float s1 = 0.f, s2 = 0.f;
  for (long i = threadIdx.x; i < N; i += blockDim.x) {
    long bi = i / chw, pix = i - bi * chw;
    float v = cur[bi * bstride + (long)c * chw + pix];
    s1 += v; s2 += v * v;
  }
  __shared__ float r1[256], r2[256];
  r1[threadIdx.x] = s1; r2[threadIdx.x] = s2;
  __syncthreads();
  for (int off = 128; off > 0; off >>= 1) {
    if (threadIdx.x < off) {
      r1[threadIdx.x] += r1[threadIdx.x + off];
      r2[threadIdx.x] += r2[threadIdx.x + off];
    }
    __syncthreads();
  }
  if (threadIdx.x == 0) {
    float mean = r1[0] / (float)N;
    float var  = r2[0] / (float)N - mean * mean;
    float sc   = g[c] * rsqrtf(var + BN_EPS);
    ss[c]       = sc;
    ss[256 + c] = beta[c] - mean * sc;
  }
}

// ---------------------------------------------------------------------------
// Fused BN + LIF: mem = beta*mem + bn(cur); spk = mem>=1; mem -= spk.
// Spikes stored as f16 (exact 0/1) to halve gather bytes in the next conv;
// integer spike counting keeps the rate deterministic across graph replays.
// ---------------------------------------------------------------------------
__global__ void bn_lif(const float* __restrict__ cur, float* __restrict__ mem,
                       _Float16* __restrict__ spk_out,
                       const float* __restrict__ ss, float* __restrict__ acc,
                       unsigned int* __restrict__ spk_count,
                       long total, int C, long chw) {
  const long stride = (long)gridDim.x * blockDim.x;
  int cnt = 0;
  for (long i = (long)blockIdx.x * blockDim.x + threadIdx.x; i < total; i += stride) {
    int c = (int)((i / chw) % C);
    float y = cur[i] * ss[c] + ss[256 + c];
    float m = LIF_BETA * mem[i] + y;
    int fired = (m >= 1.0f);
    float spk = fired ? 1.0f : 0.0f;
    mem[i] = m - spk;
    spk_out[i] = (_Float16)spk;
    if (acc) acc[i] += spk;
    cnt += fired;
  }
  __shared__ int red[256];
  red[threadIdx.x] = cnt;
  __syncthreads();
  for (int off = 128; off > 0; off >>= 1) {
    if (threadIdx.x < off) red[threadIdx.x] += red[threadIdx.x + off];
    __syncthreads();
  }
  if (threadIdx.x == 0 && red[0]) atomicAdd(spk_count, (unsigned)red[0]);
}

// sum[b,c,y,x] = lat3[b,c,y,x] + lat4[b,c,(y*Hi)/Ho,(x*Wi)/Wo]  (torch nearest)
__global__ void up_add(const float* __restrict__ lat3, const float* __restrict__ lat4,
                       float* __restrict__ sum, int B, int C, int Ho, int Wo,
                       int Hi, int Wi) {
  long total = (long)B * C * Ho * Wo;
  long i = (long)blockIdx.x * blockDim.x + threadIdx.x;
  if (i >= total) return;
  int x = (int)(i % Wo); long t = i / Wo;
  int y = (int)(t % Ho); t /= Ho;
  int c = (int)(t % C);  int b = (int)(t / C);
  int ih = (y * Hi) / Ho, iw = (x * Wi) / Wo;
  sum[i] = lat3[i] + lat4[(((long)b * C + c) * Hi + ih) * Wi + iw];
}

__global__ void finish_rate(const unsigned int* __restrict__ spk_count,
                            float* __restrict__ out, float denom_inv) {
  if (threadIdx.x == 0 && blockIdx.x == 0)
    out[0] = (float)(*spk_count) * denom_inv;
}

// ---------------------------------------------------------------------------
extern "C" void kernel_launch(void* const* d_in, const int* in_sizes, int n_in,
                              void* d_out, int out_size, void* d_ws, size_t ws_size,
                              hipStream_t stream) {
  (void)in_sizes; (void)n_in; (void)out_size; (void)ws_size;

  const float* x = (const float*)d_in[0];
  const float* conv_w[4] = {(const float*)d_in[1], (const float*)d_in[2],
                            (const float*)d_in[3], (const float*)d_in[4]};
  const float* bn_g[4] = {(const float*)d_in[5], (const float*)d_in[7],
                          (const float*)d_in[9], (const float*)d_in[11]};
  const float* bn_b[4] = {(const float*)d_in[6], (const float*)d_in[8],
                          (const float*)d_in[10], (const float*)d_in[12]};
  const float* lat3_w = (const float*)d_in[13]; const float* lat3_b = (const float*)d_in[14];
  const float* lat4_w = (const float*)d_in[15]; const float* lat4_b = (const float*)d_in[16];
  const float* fpn3_w = (const float*)d_in[17]; const float* fpn3_b = (const float*)d_in[18];
  // p3: 19..26  p4: 27..34  (cls1_w,cls1_b,cls2_w,cls2_b,reg1_w,reg1_b,reg2_w,reg2_b)

  const int B = 2, T = 10;
  const int H1 = 260, W1 = 346, H2 = 130, W2 = 173, H3 = 65, W3 = 87, H4 = 33, W4 = 44;
  const long P1 = (long)H1 * W1, P2 = (long)H2 * W2, P3 = (long)H3 * W3, P4 = (long)H4 * W4;
  const long n1 = (long)B * 32 * P1, n2 = (long)B * 64 * P2;
  const long n3 = (long)B * 128 * P3, n4 = (long)B * 256 * P4;

  // ---- workspace bump allocator (256B aligned) ----
  char* bp = (char*)d_ws;
  size_t off = 0;
  auto alloc = [&](size_t bytes) -> char* {
    off = (off + 255) & ~(size_t)255;
    char* p = bp + off;
    off += bytes;
    return p;
  };
  float* cur1 = (float*)alloc(n1 * 4);  float* cur2 = (float*)alloc(n2 * 4);
  float* cur3 = (float*)alloc(n3 * 4);  float* cur4 = (float*)alloc(n4 * 4);
  float* mem1 = (float*)alloc(n1 * 4);  float* mem2 = (float*)alloc(n2 * 4);
  float* mem3 = (float*)alloc(n3 * 4);  float* mem4 = (float*)alloc(n4 * 4);
  _Float16* s1 = (_Float16*)alloc(n1 * 2);  _Float16* s2 = (_Float16*)alloc(n2 * 2);
  _Float16* s3 = (_Float16*)alloc(n3 * 2);  _Float16* s4 = (_Float16*)alloc(n4 * 2);
  float* a3   = (float*)alloc(n3 * 4);  float* a4   = (float*)alloc(n4 * 4);
  float* lat3buf = (float*)alloc((size_t)B * 128 * P3 * 4);
  float* lat4buf = (float*)alloc((size_t)B * 128 * P4 * 4);
  float* sum3    = (float*)alloc((size_t)B * 128 * P3 * 4);
  float* fpn3buf = (float*)alloc((size_t)B * 128 * P3 * 4);
  float* hbuf    = (float*)alloc((size_t)B * 128 * P3 * 4);
  float* ssbuf   = (float*)alloc(512 * 4);
  unsigned int* spcnt = (unsigned int*)alloc(64);

  auto allocH = [&](size_t elems) -> _Float16* { return (_Float16*)alloc(elems * 2); };
  _Float16* pkc1   = allocH((size_t)2 * 1 * 512);    // 32 x 18
  _Float16* pkc2   = allocH((size_t)4 * 9 * 512);    // 64 x 288
  _Float16* pkc3   = allocH((size_t)8 * 18 * 512);   // 128 x 576
  _Float16* pkc4   = allocH((size_t)16 * 36 * 512);  // 256 x 1152
  _Float16* pklat3 = allocH((size_t)8 * 4 * 512);    // 128 x 128
  _Float16* pklat4 = allocH((size_t)8 * 8 * 512);    // 128 x 256
  _Float16* pkfpn3 = allocH((size_t)8 * 36 * 512);   // 128 x 1152
  _Float16* pkp3c1 = allocH((size_t)8 * 36 * 512);
  _Float16* pkp3c2 = allocH((size_t)1 * 4 * 512);    // 6 x 128
  _Float16* pkp3r1 = allocH((size_t)8 * 36 * 512);
  _Float16* pkp3r2 = allocH((size_t)1 * 4 * 512);    // 12 x 128
  _Float16* pkp4c1 = allocH((size_t)8 * 36 * 512);
  _Float16* pkp4c2 = allocH((size_t)1 * 4 * 512);
  _Float16* pkp4r1 = allocH((size_t)8 * 36 * 512);
  _Float16* pkp4r2 = allocH((size_t)1 * 4 * 512);

  auto launch_pack = [&](const float* w, _Float16* wpk, int Cout, int nK) {
    int nkc = (nK + 31) / 32;
    int total = ((Cout + 15) / 16) * nkc * 512;
    pack_weights<<<(total + 255) / 256, 256, 0, stream>>>(w, wpk, Cout, nK, nkc);
  };
  launch_pack(conv_w[0], pkc1, 32, 18);
  launch_pack(conv_w[1], pkc2, 64, 288);
  launch_pack(conv_w[2], pkc3, 128, 576);
  launch_pack(conv_w[3], pkc4, 256, 1152);
  launch_pack(lat3_w, pklat3, 128, 128);
  launch_pack(lat4_w, pklat4, 128, 256);
  launch_pack(fpn3_w, pkfpn3, 128, 1152);
  launch_pack((const float*)d_in[19], pkp3c1, 128, 1152);
  launch_pack((const float*)d_in[21], pkp3c2, 6, 128);
  launch_pack((const float*)d_in[23], pkp3r1, 128, 1152);
  launch_pack((const float*)d_in[25], pkp3r2, 12, 128);
  launch_pack((const float*)d_in[27], pkp4c1, 128, 1152);
  launch_pack((const float*)d_in[29], pkp4c2, 6, 128);
  launch_pack((const float*)d_in[31], pkp4r1, 128, 1152);
  launch_pack((const float*)d_in[33], pkp4r2, 12, 128);

  // zero recurrent state / accumulators each call (d_ws is not re-poisoned)
  hipMemsetAsync(mem1, 0, n1 * 4, stream);
  hipMemsetAsync(mem2, 0, n2 * 4, stream);
  hipMemsetAsync(mem3, 0, n3 * 4, stream);
  hipMemsetAsync(mem4, 0, n4 * 4, stream);
  hipMemsetAsync(a3, 0, n3 * 4, stream);
  hipMemsetAsync(a4, 0, n4 * 4, stream);
  hipMemsetAsync(spcnt, 0, sizeof(unsigned int), stream);

  auto gridFor = [](long n) { return (unsigned)((n + 1023) / 1024); };

  // ---- recurrent backbone over T timesteps ----
  for (int t = 0; t < T; ++t) {
    const float* x_t = x + (size_t)t * 2 * P1;                 // x[b][t][c][h][w]
    conv_dispatch<float>(x_t, pkc1, nullptr, cur1, B, 2, H1, W1, (long)T * 2 * P1,
                         32, H1, W1, 3, 1, 1, 1.f, 0, stream);
    bn_stats<<<32, 256, 0, stream>>>(cur1, bn_g[0], bn_b[0], ssbuf, 32, B, P1);
    bn_lif<<<gridFor(n1), 256, 0, stream>>>(cur1, mem1, s1, ssbuf, nullptr, spcnt, n1, 32, P1);

    conv_dispatch<_Float16>(s1, pkc2, nullptr, cur2, B, 32, H1, W1, 32 * P1,
                            64, H2, W2, 3, 2, 1, 1.f, 0, stream);
    bn_stats<<<64, 256, 0, stream>>>(cur2, bn_g[1], bn_b[1], ssbuf, 64, B, P2);
    bn_lif<<<gridFor(n2), 256, 0, stream>>>(cur2, mem2, s2, ssbuf, nullptr, spcnt, n2, 64, P2);

    conv_dispatch<_Float16>(s2, pkc3, nullptr, cur3, B, 64, H2, W2, 64 * P2,
                            128, H3, W3, 3, 2, 1, 1.f, 0, stream);
    bn_stats<<<128, 256, 0, stream>>>(cur3, bn_g[2], bn_b[2], ssbuf, 128, B, P3);
    bn_lif<<<gridFor(n3), 256, 0, stream>>>(cur3, mem3, s3, ssbuf, a3, spcnt, n3, 128, P3);

    conv_dispatch<_Float16>(s3, pkc4, nullptr, cur4, B, 128, H3, W3, 128 * P3,
                            256, H4, W4, 3, 2, 1, 1.f, 0, stream);
    bn_stats<<<256, 256, 0, stream>>>(cur4, bn_g[3], bn_b[3], ssbuf, 256, B, P4);
    bn_lif<<<gridFor(n4), 256, 0, stream>>>(cur4, mem4, s4, ssbuf, a4, spcnt, n4, 256, P4);
  }

  // ---- FPN (f = a/T folded into in_scale = 0.1) ----
  conv_dispatch<float>(a4, pklat4, lat4_b, lat4buf, B, 256, H4, W4, 256 * P4,
                       128, H4, W4, 1, 1, 0, 0.1f, 0, stream);
  conv_dispatch<float>(a3, pklat3, lat3_b, lat3buf, B, 128, H3, W3, 128 * P3,
                       128, H3, W3, 1, 1, 0, 0.1f, 0, stream);
  long tot3 = (long)B * 128 * P3;
  up_add<<<(unsigned)((tot3 + 255) / 256), 256, 0, stream>>>(lat3buf, lat4buf, sum3,
                                                             B, 128, H3, W3, H4, W4);
  conv_dispatch<float>(sum3, pkfpn3, fpn3_b, fpn3buf, B, 128, H3, W3, 128 * P3,
                       128, H3, W3, 3, 1, 1, 1.f, 0, stream);

  // ---- heads -> d_out (cls3 | box3 | cls4 | box4 | rate) ----
  float* outp = (float*)d_out;
  const long o_box3 = (long)B * 6 * P3;                        //  67,860
  const long o_cls4 = o_box3 + (long)B * 12 * P3;              // 203,580
  const long o_box4 = o_cls4 + (long)B * 6 * P4;               // 221,004
  const long o_rate = o_box4 + (long)B * 12 * P4;              // 255,852

  conv_dispatch<float>(fpn3buf, pkp3c1, (const float*)d_in[20], hbuf, B, 128, H3, W3,
                       128 * P3, 128, H3, W3, 3, 1, 1, 1.f, 1, stream);
  conv_dispatch<float>(hbuf, pkp3c2, (const float*)d_in[22], outp, B, 128, H3, W3,
                       128 * P3, 6, H3, W3, 1, 1, 0, 1.f, 0, stream);
  conv_dispatch<float>(fpn3buf, pkp3r1, (const float*)d_in[24], hbuf, B, 128, H3, W3,
                       128 * P3, 128, H3, W3, 3, 1, 1, 1.f, 1, stream);
  conv_dispatch<float>(hbuf, pkp3r2, (const float*)d_in[26], outp + o_box3, B, 128, H3, W3,
                       128 * P3, 12, H3, W3, 1, 1, 0, 1.f, 0, stream);

  conv_dispatch<float>(lat4buf, pkp4c1, (const float*)d_in[28], hbuf, B, 128, H4, W4,
                       128 * P4, 128, H4, W4, 3, 1, 1, 1.f, 1, stream);
  conv_dispatch<float>(hbuf, pkp4c2, (const float*)d_in[30], outp + o_cls4, B, 128, H4, W4,
                       128 * P4, 6, H4, W4, 1, 1, 0, 1.f, 0, stream);
  conv_dispatch<float>(lat4buf, pkp4r1, (const float*)d_in[32], hbuf, B, 128, H4, W4,
                       128 * P4, 128, H4, W4, 3, 1, 1, 1.f, 1, stream);
  conv_dispatch<float>(hbuf, pkp4r2, (const float*)d_in[34], outp + o_box4, B, 128, H4, W4,
                       128 * P4, 12, H4, W4, 1, 1, 0, 1.f, 0, stream);

  // rate = sp / (neurons_per_step * T * T);  neurons_per_step = 10,827,264
  finish_rate<<<1, 1, 0, stream>>>(spcnt, outp + o_rate, 1.0f / (10827264.0f * 100.0f));
}
